// GPT2_57586921504864
// MI455X (gfx1250) — matrix operand order repaired and checked by reading, not compiled, not run on hardware
//
#include <hip/hip_runtime.h>
#include <hip/hip_bf16.h>

typedef __bf16 bf16;
typedef __attribute__((ext_vector_type(16))) __bf16 v16bf;
typedef __attribute__((ext_vector_type(8)))  __bf16 v8bf;
typedef __attribute__((ext_vector_type(8)))  float  v8f;

#define NLAYER 12
#define NHEAD  12
#define EMB    768
#define HDIM   64
#define FFDIM  3072
#define NVOCAB 50257
#define BATCH  2
#define SEQ    1024
#define NTOK   (BATCH*SEQ)

// gfx1250 async global->LDS path (guarded: falls back to VGPR staging if the
// toolchain doesn't declare the builtins).  The builtins take pointers to a
// 16-byte int vector in AS1 (global) / AS3 (LDS) per the clang signature.
#if defined(__AMDGCN__) && __has_builtin(__builtin_amdgcn_global_load_async_to_lds_b128) && __has_builtin(__builtin_amdgcn_s_wait_asynccnt)
#define USE_ASYNC 1
typedef int v4i __attribute__((vector_size(16)));
typedef __attribute__((address_space(1))) v4i gv4i;
typedef __attribute__((address_space(3))) v4i lv4i;
#else
#define USE_ASYNC 0
#endif

// ---------------------------------------------------------------------------
// Fragment load helpers (wave32 WMMA layouts, CDNA5 ISA §7.12.2)
// ---------------------------------------------------------------------------

// 16 contiguous bf16 from p (16B-aligned) -> element e holds p[e].
__device__ inline v16bf ld16(const bf16* p) {
  v8bf a = *(const v8bf*)p;
  v8bf b = *(const v8bf*)(p + 8);
  v16bf r;
#pragma unroll
  for (int i = 0; i < 8; ++i) { r[i] = a[i]; r[i + 8] = b[i]; }
  return r;
}

// A-matrix 16x32 bf16 fragment, per-lane: elements 0-7 = rowp[abase..abase+7],
// elements 8-15 = rowp[abase+16..abase+23]; abase = (lane<16) ? 0 : 8.
__device__ inline v16bf ldA(const bf16* rowp, int abase) {
  v8bf a = *(const v8bf*)(rowp + abase);
  v8bf b = *(const v8bf*)(rowp + abase + 16);
  v16bf r;
#pragma unroll
  for (int i = 0; i < 8; ++i) { r[i] = a[i]; r[i + 8] = b[i]; }
  return r;
}

__device__ inline float gelu_tanh(float x) {
  const float c = 0.7978845608028654f;
  float t = tanhf(c * (x + 0.044715f * x * x * x));
  return 0.5f * x * (1.f + t);
}

// ---------------------------------------------------------------------------
// Embedding: h[t,:] = tok_emb[idx[t],:] + pos_emb[t%SEQ,:]
// ---------------------------------------------------------------------------
__global__ __launch_bounds__(256) void k_embed(const int* __restrict__ idx,
                                               const float* __restrict__ tok,
                                               const float* __restrict__ pos,
                                               float* __restrict__ h) {
  int t = blockIdx.x;
  int s = t % SEQ;
  int id = idx[t];
  for (int e = threadIdx.x; e < EMB; e += 256)
    h[(size_t)t * EMB + e] = tok[(size_t)id * EMB + e] + pos[(size_t)s * EMB + e];
}

// ---------------------------------------------------------------------------
// LayerNorm (fp32 in) -> bf16 out.  One block (256 thr, 8 waves) per row.
// ---------------------------------------------------------------------------
__device__ inline float block_sum(float x, float* red) {
#pragma unroll
  for (int m = 16; m >= 1; m >>= 1) x += __shfl_xor(x, m);
  int wid = threadIdx.x >> 5, lane = threadIdx.x & 31;
  if (lane == 0) red[wid] = x;
  __syncthreads();
  if (threadIdx.x == 0) {
    float t = red[0];
    for (int i = 1; i < 8; ++i) t += red[i];
    red[0] = t;
  }
  __syncthreads();
  float r = red[0];
  __syncthreads();
  return r;
}

__global__ __launch_bounds__(256) void k_ln(const float* __restrict__ hsrc,
                                            const float* __restrict__ g,
                                            const float* __restrict__ bia,
                                            bf16* __restrict__ out) {
  __shared__ float red[8];
  int t = blockIdx.x;
  const float* row = hsrc + (size_t)t * EMB;
  float v[3];
  float s = 0.f;
#pragma unroll
  for (int i = 0; i < 3; ++i) { v[i] = row[threadIdx.x + 256 * i]; s += v[i]; }
  float mean = block_sum(s, red) * (1.f / EMB);
  float s2 = 0.f;
#pragma unroll
  for (int i = 0; i < 3; ++i) { float d = v[i] - mean; s2 += d * d; }
  float var = block_sum(s2, red) * (1.f / EMB);
  float rstd = rsqrtf(var + 1e-5f);
#pragma unroll
  for (int i = 0; i < 3; ++i) {
    int e = threadIdx.x + 256 * i;
    out[(size_t)t * EMB + e] = (bf16)((v[i] - mean) * rstd * g[e] + bia[e]);
  }
}

// ---------------------------------------------------------------------------
// GEMM: C[M,N] = A[M,K](bf16) x W[K,N](fp32 -> bf16) ; epilogue bias/gelu/resid
// N, K are compile-time so strided W loads use immediate offsets.
// Block 256 thr (8 waves), tile 128x128x32; wave = 64 rows x 32 cols = 8 WMMAs.
// Software pipelined with double-buffered LDS: async A-copy (gfx1250
// global_load_async_to_lds) and fp32 W loads for tile i+1 run under the
// WMMAs of tile i; one barrier per k-step.
// ---------------------------------------------------------------------------
template<int N, int K>
__global__ __launch_bounds__(256) void k_gemm(const bf16* __restrict__ A,
                                              const float* __restrict__ W,
                                              const float* __restrict__ bias,
                                              const float* __restrict__ resid,
                                              float* __restrict__ outF,
                                              bf16* __restrict__ outB,
                                              int act) {
  __shared__ bf16 As[2][128][40];  // [buf][m][k], +8 pad
  __shared__ bf16 Bs[2][128][40];  // [buf][n][k] (W^T), +8 pad
  int n0 = blockIdx.x * 128;
  int m0 = blockIdx.y * 128;
  int tid = threadIdx.x;
  int lane = tid & 31;
  int l15 = lane & 15;
  int upper = (lane >= 16);
  int abase = upper ? 8 : 0;
  int wave = tid >> 5;
  int wrow = (wave & 1) * 64;        // 2 wave-rows
  int wcol = (wave >> 1) * 32;       // 4 wave-cols

  v8f acc[4][2];
#pragma unroll
  for (int rt = 0; rt < 4; ++rt)
#pragma unroll
    for (int ct = 0; ct < 2; ++ct)
#pragma unroll
      for (int r = 0; r < 8; ++r) acc[rt][ct][r] = 0.f;

  int arow = tid >> 1, acol = (tid & 1) * 16;   // A-tile: 16 bf16 per thread
  int wn  = tid >> 1;                           // W-tile column within tile
  int wk0 = (tid & 1) * 16;                     // k sub-range 0..15 / 16..31
  int ncol = n0 + wn;
  bool colok = (ncol < N);

  const bf16* aptr = &A[(size_t)(m0 + arow) * K + acol];

  // issue A-tile copy for k-offset k0 into LDS buffer b
  auto issueA = [&](int k0, int b) {
    const bf16* ap = aptr + k0;
#if USE_ASYNC
    bf16* lp = &As[b][arow][acol];
    __builtin_amdgcn_global_load_async_to_lds_b128((gv4i*)ap, (lv4i*)lp, 0, 0);
    __builtin_amdgcn_global_load_async_to_lds_b128((gv4i*)(ap + 8),
                                                   (lv4i*)(lp + 8), 0, 0);
#else
    *(v8bf*)&As[b][arow][acol]     = *(const v8bf*)ap;
    *(v8bf*)&As[b][arow][acol + 8] = *(const v8bf*)(ap + 8);
#endif
  };

  // W column strip for the next tile lands in registers (overlaps compute)
  float wv[16];
  const float* wp = colok ? &W[(size_t)wk0 * N + ncol] : nullptr;
  auto loadW = [&]() {
    if (colok) {
#pragma unroll
      for (int j = 0; j < 16; ++j) wv[j] = wp[(size_t)j * N];  // imm offsets
      __builtin_prefetch(wp + (size_t)64 * N, 0, 1);           // tile i+2
      wp += (size_t)32 * N;
    } else {
#pragma unroll
      for (int j = 0; j < 16; ++j) wv[j] = 0.f;
    }
  };

  issueA(0, 0);
  loadW();

  constexpr int T = K / 32;
  for (int i = 0; i < T; ++i) {
    int bsel = i & 1;
    // publish tile i: convert staged W regs -> LDS, wait async A, barrier
    v8bf blo, bhi;
#pragma unroll
    for (int j = 0; j < 8; ++j) { blo[j] = (bf16)wv[j]; bhi[j] = (bf16)wv[8 + j]; }
    *(v8bf*)&Bs[bsel][wn][wk0]     = blo;
    *(v8bf*)&Bs[bsel][wn][wk0 + 8] = bhi;
#if USE_ASYNC
    __builtin_amdgcn_s_wait_asynccnt(0);
#endif
    __syncthreads();

    // start tile i+1 transfers (in flight under the WMMAs below)
    if (i + 1 < T) {
      issueA((i + 1) * 32, bsel ^ 1);
      loadW();
    }

    // ---- WMMA k-step on buffer bsel ----------------------------------
    v16bf bq[2];
#pragma unroll
    for (int ct = 0; ct < 2; ++ct)
      bq[ct] = ldA(&Bs[bsel][wcol + ct * 16 + l15][0], abase);
#pragma unroll
    for (int rt = 0; rt < 4; ++rt) {
      v16bf aq = ldA(&As[bsel][wrow + rt * 16 + l15][0], abase);
      acc[rt][0] = __builtin_amdgcn_wmma_f32_16x16x32_bf16(
          false, aq, false, bq[0], (short)0, acc[rt][0], false, false);
      acc[rt][1] = __builtin_amdgcn_wmma_f32_16x16x32_bf16(
          false, aq, false, bq[1], (short)0, acc[rt][1], false, false);
    }
  }

  // epilogue: C/D layout -> lane half selects rows 0-7 / 8-15, col = lane&15
#pragma unroll
  for (int rt = 0; rt < 4; ++rt)
#pragma unroll
    for (int ct = 0; ct < 2; ++ct)
#pragma unroll
      for (int r = 0; r < 8; ++r) {
        int m = m0 + wrow + rt * 16 + (upper ? 8 : 0) + r;
        int n = n0 + wcol + ct * 16 + l15;
        if (n < N) {
          float val = acc[rt][ct][r];
          if (bias)  val += bias[n];
          if (act)   val = gelu_tanh(val);
          if (resid) val += resid[(size_t)m * N + n];
          if (outF)  outF[(size_t)m * N + n] = val;
          if (outB)  outB[(size_t)m * N + n] = (bf16)val;
        }
      }
}

// ---------------------------------------------------------------------------
// Causal flash attention. grid = (B*NH, SEQ/64), block = 128 (4 waves).
// Each wave owns a 16-query strip; keys processed in tiles of 32.
// q/k/v/o layout: [B, S, NH*HDIM] bf16.
// ---------------------------------------------------------------------------
__global__ __launch_bounds__(128) void k_attn(const bf16* __restrict__ q,
                                              const bf16* __restrict__ k,
                                              const bf16* __restrict__ v,
                                              bf16* __restrict__ o) {
  __shared__ bf16 plds[4][16][32];   // per-wave P tile (16 q x 32 keys)
  int bh = blockIdx.x;
  int b  = bh / NHEAD;
  int hh = bh % NHEAD;
  int wave = threadIdx.x >> 5;
  int lane = threadIdx.x & 31;
  int l15 = lane & 15;
  int upper = (lane >= 16);
  int abase = upper ? 8 : 0;
  int q0 = blockIdx.y * 64 + wave * 16;

  const bf16* qbase = q + (size_t)b * SEQ * EMB + hh * HDIM;
  const bf16* kbase = k + (size_t)b * SEQ * EMB + hh * HDIM;
  const bf16* vbase = v + (size_t)b * SEQ * EMB + hh * HDIM;

  // Q strip as two A fragments (d = 0..31, 32..63)
  const bf16* qrow = qbase + (size_t)(q0 + l15) * EMB;
  v16bf qa0 = ldA(qrow, abase);
  v16bf qa1 = ldA(qrow + 32, abase);

  v8f oacc[4];
#pragma unroll
  for (int g = 0; g < 4; ++g)
#pragma unroll
    for (int r = 0; r < 8; ++r) oacc[g][r] = 0.f;
  float m_i[8], l_i[8];
#pragma unroll
  for (int r = 0; r < 8; ++r) { m_i[r] = -1e30f; l_i[r] = 0.f; }

  const float scale = 0.125f;   // 1/sqrt(64)
  int kend = q0 + 16;           // causal bound (exclusive)

  for (int kt = 0; kt < kend; kt += 32) {
    v8f sc[2];
#pragma unroll
    for (int sub = 0; sub < 2; ++sub) {
      const bf16* krow = kbase + (size_t)(kt + sub * 16 + l15) * EMB;
      v16bf kb0 = ldA(krow, abase);         // d 0..31 chunk
      v16bf kb1 = ldA(krow + 32, abase);    // d 32..63 chunk
      v8f z;
#pragma unroll
      for (int r = 0; r < 8; ++r) z[r] = 0.f;
      v8f s0 = __builtin_amdgcn_wmma_f32_16x16x32_bf16(
          false, qa0, false, kb0, (short)0, z, false, false);
      s0 = __builtin_amdgcn_wmma_f32_16x16x32_bf16(
          false, qa1, false, kb1, (short)0, s0, false, false);
      int key = kt + sub * 16 + l15;
#pragma unroll
      for (int r = 0; r < 8; ++r) {
        int qi = q0 + (upper ? 8 : 0) + r;
        s0[r] = (key <= qi) ? s0[r] * scale : -1e30f;
      }
      sc[sub] = s0;
    }

    // row max over 32 keys (both subs + 16 lanes of each half-wave)
    float mx[8];
#pragma unroll
    for (int r = 0; r < 8; ++r) mx[r] = fmaxf(sc[0][r], sc[1][r]);
#pragma unroll
    for (int msk = 1; msk <= 8; msk <<= 1)
#pragma unroll
      for (int r = 0; r < 8; ++r) mx[r] = fmaxf(mx[r], __shfl_xor(mx[r], msk));

    float alpha[8];
#pragma unroll
    for (int r = 0; r < 8; ++r) {
      float mn = fmaxf(m_i[r], mx[r]);
      alpha[r] = __expf(m_i[r] - mn);
      m_i[r] = mn;
    }
    float rsum[8];
#pragma unroll
    for (int r = 0; r < 8; ++r) rsum[r] = 0.f;
#pragma unroll
    for (int sub = 0; sub < 2; ++sub)
#pragma unroll
      for (int r = 0; r < 8; ++r) {
        float p = __expf(sc[sub][r] - m_i[r]);
        sc[sub][r] = p;
        rsum[r] += p;
      }
#pragma unroll
    for (int msk = 1; msk <= 8; msk <<= 1)
#pragma unroll
      for (int r = 0; r < 8; ++r) rsum[r] += __shfl_xor(rsum[r], msk);
#pragma unroll
    for (int r = 0; r < 8; ++r) l_i[r] = l_i[r] * alpha[r] + rsum[r];
#pragma unroll
    for (int g = 0; g < 4; ++g)
#pragma unroll
      for (int r = 0; r < 8; ++r) oacc[g][r] *= alpha[r];

    // P: C-layout -> LDS -> A-layout (same-wave DS ops are in-order)
#pragma unroll
    for (int sub = 0; sub < 2; ++sub)
#pragma unroll
      for (int r = 0; r < 8; ++r)
        plds[wave][(upper ? 8 : 0) + r][sub * 16 + l15] = (bf16)sc[sub][r];
    asm volatile("" ::: "memory");
    __builtin_amdgcn_wave_barrier();
    v16bf pa = ld16(&plds[wave][l15][upper ? 16 : 0]);

    // O += P x V : B fragment lane: col = d-group*16 + lane&15, K = key
#pragma unroll
    for (int g = 0; g < 4; ++g) {
      const bf16* vp = vbase + (size_t)(kt + (upper ? 16 : 0)) * EMB + g * 16 + l15;
      v16bf vfr;
#pragma unroll
      for (int e = 0; e < 16; ++e) vfr[e] = vp[(size_t)e * EMB];
      oacc[g] = __builtin_amdgcn_wmma_f32_16x16x32_bf16(
          false, pa, false, vfr, (short)0, oacc[g], false, false);
    }
  }

  // normalize and store
#pragma unroll
  for (int g = 0; g < 4; ++g)
#pragma unroll
    for (int r = 0; r < 8; ++r) {
      int row = q0 + (upper ? 8 : 0) + r;
      o[(size_t)(b * SEQ + row) * EMB + hh * HDIM + g * 16 + l15] =
          (bf16)(oacc[g][r] / l_i[r]);
    }
}

// ---------------------------------------------------------------------------
// Host-side orchestration
// ---------------------------------------------------------------------------
template<int N, int K>
static inline void launch_gemm(hipStream_t st, const bf16* A, const float* W,
                               const float* bias, const float* resid,
                               float* outF, bf16* outB, int act) {
  dim3 grid((N + 127) / 128, NTOK / 128);
  k_gemm<N, K><<<grid, 256, 0, st>>>(A, W, bias, resid, outF, outB, act);
}

extern "C" void kernel_launch(void* const* d_in, const int* in_sizes, int n_in,
                              void* d_out, int out_size, void* d_ws, size_t ws_size,
                              hipStream_t stream) {
  (void)in_sizes; (void)n_in; (void)out_size; (void)ws_size;
  const int*   indices = (const int*)  d_in[0];
  const float* tok_emb = (const float*)d_in[1];
  const float* pos_emb = (const float*)d_in[2];
  const float* ln1_g   = (const float*)d_in[3];
  const float* ln1_b   = (const float*)d_in[4];
  const float* wq      = (const float*)d_in[5];
  const float* bq      = (const float*)d_in[6];
  const float* wk      = (const float*)d_in[7];
  const float* bk      = (const float*)d_in[8];
  const float* wv      = (const float*)d_in[9];
  const float* bv      = (const float*)d_in[10];
  const float* wo      = (const float*)d_in[11];
  const float* bo      = (const float*)d_in[12];
  const float* ln2_g   = (const float*)d_in[13];
  const float* ln2_b   = (const float*)d_in[14];
  const float* w1      = (const float*)d_in[15];
  const float* b1      = (const float*)d_in[16];
  const float* w2      = (const float*)d_in[17];
  const float* b2      = (const float*)d_in[18];
  const float* lnf_g   = (const float*)d_in[19];
  const float* lnf_b   = (const float*)d_in[20];
  const float* w_out   = (const float*)d_in[21];

  char* p = (char*)d_ws;
  float* h  = (float*)p; p += (size_t)NTOK * EMB * 4;
  bf16* x   = (bf16*)p;  p += (size_t)NTOK * EMB * 2;
  bf16* qb  = (bf16*)p;  p += (size_t)NTOK * EMB * 2;
  bf16* kb  = (bf16*)p;  p += (size_t)NTOK * EMB * 2;
  bf16* vb  = (bf16*)p;  p += (size_t)NTOK * EMB * 2;
  bf16* ob  = (bf16*)p;  p += (size_t)NTOK * EMB * 2;
  bf16* ffb = (bf16*)p;  p += (size_t)NTOK * FFDIM * 2;

  k_embed<<<NTOK, 256, 0, stream>>>(indices, tok_emb, pos_emb, h);

  for (int l = 0; l < NLAYER; ++l) {
    size_t wofs = (size_t)l * EMB * EMB;
    k_ln<<<NTOK, 256, 0, stream>>>(h, ln1_g + l * EMB, ln1_b + l * EMB, x);
    launch_gemm<EMB, EMB>(stream, x, wq + wofs, bq + l * EMB, nullptr, nullptr, qb, 0);
    launch_gemm<EMB, EMB>(stream, x, wk + wofs, bk + l * EMB, nullptr, nullptr, kb, 0);
    launch_gemm<EMB, EMB>(stream, x, wv + wofs, bv + l * EMB, nullptr, nullptr, vb, 0);
    k_attn<<<dim3(BATCH * NHEAD, SEQ / 64), 128, 0, stream>>>(qb, kb, vb, ob);
    launch_gemm<EMB, EMB>(stream, ob, wo + wofs, bo + l * EMB, h, h, nullptr, 0);
    k_ln<<<NTOK, 256, 0, stream>>>(h, ln2_g + l * EMB, ln2_b + l * EMB, x);
    launch_gemm<FFDIM, EMB>(stream, x, w1 + (size_t)l * EMB * FFDIM, b1 + l * FFDIM,
                            nullptr, nullptr, ffb, 1);
    launch_gemm<EMB, FFDIM>(stream, ffb, w2 + (size_t)l * FFDIM * EMB, b2 + l * EMB,
                            h, h, nullptr, 0);
  }

  k_ln<<<NTOK, 256, 0, stream>>>(h, lnf_g, lnf_b, x);
  launch_gemm<NVOCAB, EMB>(stream, x, w_out, nullptr, nullptr, (float*)d_out, nullptr, 0);
}
